// CLI_v1_63702954934484
// MI455X (gfx1250) — matrix operand order repaired and checked by reading, not compiled, not run on hardware
//
#include <hip/hip_runtime.h>
#include <math.h>

typedef __attribute__((ext_vector_type(16))) __bf16 v16bf;
typedef __attribute__((ext_vector_type(8)))  float  v8f;

#define DD   512
#define NPTS 2048
#define NBAT 16

// ---------------------------------------------------------------------------
// Kernel 1: build bf16 transposed weight blocks.
//   WbotT [n][k]  = W[512+k][n]            (for pa = a @ W_bot)
//   WdiffT[n][k]  = W[k][n] - W[512+k][n]  (for pb = b @ (W_top - W_bot))
// N-major layout => B-fragment per lane is 16 contiguous bf16 (32B, aligned).
// ---------------------------------------------------------------------------
__global__ __launch_bounds__(256)
void prep_w(const float* __restrict__ W,
            unsigned short* __restrict__ wbot_u,
            unsigned short* __restrict__ wdiff_u) {
  __bf16* WbotT  = (__bf16*)wbot_u;
  __bf16* WdiffT = (__bf16*)wdiff_u;
  int idx = blockIdx.x * 256 + threadIdx.x;   // 0 .. 512*512-1
  int n = idx >> 9;
  int k = idx & (DD - 1);
  float wt = W[(size_t)k * DD + n];
  float wb = W[(size_t)(DD + k) * DD + n];
  WbotT [(size_t)n * DD + k] = (__bf16)wb;
  WdiffT[(size_t)n * DD + k] = (__bf16)(wt - wb);
}

// ---------------------------------------------------------------------------
// Kernel 2/3: C[M x 512] = A_f32[M x 512] @ W^T_bf16, fp32 accumulate.
// Block = 256 threads = 8 waves; block tile 128(M) x 64(N).
// The block's 64x512 bf16 W-slice (64 KB) is staged in LDS ONCE via the
// gfx1250 async copy path (global_load_async_to_lds_b128 / ASYNCcnt), then
// each wave's B fragments come from LDS (ds_load_b128, 32B aligned).
// The LDS VDST offset is derived from the array's own address (ISA 10.2:
// generic LDS pointer carries the LDS byte offset in addr[31:0]) — this both
// gives the correct async-copy destination AND escapes the LDS object so the
// compiler cannot fold the subsequent ds reads away.
// Each wave: 16(M) x 64(N) strip = 4 WMMA accumulators; K-loop step 32.
// ---------------------------------------------------------------------------
__global__ __launch_bounds__(256)
void gemm_proj(const float* __restrict__ A,
               const unsigned short* __restrict__ WTu,
               float* __restrict__ C) {
  __shared__ __align__(64) unsigned short Bs_u[64 * DD];   // 64 KB
  const __bf16* Bs = (const __bf16*)Bs_u;

  const int lane = threadIdx.x & 31;
  const int wave = threadIdx.x >> 5;
  const int m0 = blockIdx.x * 128 + wave * 16;
  const int n0 = blockIdx.y * 64;

  // ---- Stage W-slice [n0 .. n0+63] x [0 .. 511] into LDS asynchronously ----
  {
    // addr[31:0] of a generic LDS pointer == LDS byte offset (ISA 10.2);
    // the ptrtoint also escapes Bs_u for alias analysis.
    const unsigned lds_base = (unsigned)(unsigned long long)(const void*)&Bs_u[0];
    const unsigned long long gbase =
        (unsigned long long)(const char*)WTu + (unsigned long long)n0 * DD * 2ull;
    // 64 KB = 4096 chunks of 16 B; 256 threads x 16 chunks each.
    #pragma unroll
    for (int c = 0; c < 16; ++c) {
      unsigned chunk = (unsigned)threadIdx.x + (unsigned)c * 256u;
      unsigned lds_off = lds_base + chunk * 16u;           // LDS byte address
      unsigned long long gaddr = gbase + (unsigned long long)chunk * 16ull;
      asm volatile("global_load_async_to_lds_b128 %0, %1, off"
                   :: "v"(lds_off), "v"(gaddr) : "memory");
    }
    asm volatile("s_wait_asynccnt 0x0" ::: "memory");
  }
  __syncthreads();

  const int l15   = lane & 15;
  const int hi    = lane >> 4;        // 0 for lanes 0-15, 1 for 16-31
  const int row   = m0 + l15;         // A-matrix row for this lane
  const int khalf = hi << 4;          // A frag: +16 K for upper lanes
  const int kBoff = hi << 4;          // B frag: +16 K for upper lanes

  const float* ap = A + (size_t)row * DD;

  v8f acc[4];
  #pragma unroll
  for (int nt = 0; nt < 4; ++nt)
    #pragma unroll
    for (int r = 0; r < 8; ++r) acc[nt][r] = 0.0f;

  for (int k0 = 0; k0 < DD; k0 += 32) {
    const float* a0 = ap + k0 + khalf;
    if (k0 + 32 < DD) __builtin_prefetch(ap + k0 + 32, 0, 1);  // global_prefetch_b8

    // A fragment (16x32 bf16): elems 0..15 = K base..base+15,
    // contiguous ; base = k0 + 16*(lane>=16) (matches B fragment)
    v16bf af;
    #pragma unroll
    for (int t = 0; t < 8; ++t) af[t]     = (__bf16)a0[t];
    #pragma unroll
    for (int t = 0; t < 8; ++t) af[8 + t] = (__bf16)a0[8 + t];

    // Preload all 4 B fragments from LDS (2x ds_load_b128 each), then issue
    // the 4 WMMAs back-to-back so one dscnt wait covers the batch.
    v16bf bfr[4];
    #pragma unroll
    for (int nt = 0; nt < 4; ++nt) {
      const __bf16* bp = Bs + (size_t)(nt * 16 + l15) * DD + k0 + kBoff;
      bfr[nt] = *(const v16bf*)bp;
    }
    #pragma unroll
    for (int nt = 0; nt < 4; ++nt) {
      acc[nt] = __builtin_amdgcn_wmma_f32_16x16x32_bf16(
          false, af, false, bfr[nt], (short)0, acc[nt], false, false);
    }
  }

  // C layout: VGPR r -> M = m0 + r + 8*(lane>=16), N = n0+nt*16+(lane%16)
  const int crow = m0 + (hi << 3);
  #pragma unroll
  for (int nt = 0; nt < 4; ++nt) {
    float* cp = C + (size_t)crow * DD + n0 + nt * 16 + l15;
    #pragma unroll
    for (int r = 0; r < 8; ++r) cp[(size_t)r * DD] = acc[nt][r];
  }
}

// ---------------------------------------------------------------------------
// Kernel 4: per a-point top-4 neighbors + fused output.
// Block = 256 thr = 8 waves, one wave per a-point; coords_b staged in LDS.
// key = (sq<<11)|j gives exact (dist, index) lexicographic order == top_k ties.
// ---------------------------------------------------------------------------
__global__ __launch_bounds__(256)
void fuse(const float* __restrict__ feats_a,
          const float* __restrict__ bias,
          const int*   __restrict__ coords_a,
          const int*   __restrict__ coords_b,
          const float* __restrict__ pa,
          const float* __restrict__ pb,
          float* __restrict__ out) {
  __shared__ int cbx[NPTS], cby[NPTS], cbz[NPTS];
  const int bb   = blockIdx.y;
  const int wave = threadIdx.x >> 5;
  const int lane = threadIdx.x & 31;
  const int i    = blockIdx.x * 8 + wave;

  const int* cb = coords_b + (size_t)bb * NPTS * 3;
  for (int p = threadIdx.x; p < NPTS; p += 256) {
    cbx[p] = cb[p * 3 + 0];
    cby[p] = cb[p * 3 + 1];
    cbz[p] = cb[p * 3 + 2];
  }
  __syncthreads();

  const int* ca = coords_a + ((size_t)bb * NPTS + i) * 3;
  const int ax = ca[0], ay = ca[1], az = ca[2];

  unsigned best[4] = {0xFFFFFFFFu, 0xFFFFFFFFu, 0xFFFFFFFFu, 0xFFFFFFFFu};
  for (int j = lane; j < NPTS; j += 32) {
    int dx = ax - cbx[j], dy = ay - cby[j], dz = az - cbz[j];
    unsigned sq = (unsigned)(dx * dx + dy * dy + dz * dz);  // <= 3*127^2 < 2^16
    unsigned key = (sq << 11) | (unsigned)j;
    if (key < best[3]) {
      best[3] = key;
      #pragma unroll
      for (int t = 3; t > 0; --t)
        if (best[t] < best[t - 1]) { unsigned tmp = best[t]; best[t] = best[t - 1]; best[t - 1] = tmp; }
    }
  }
  // wave32 butterfly merge of sorted-4 lists (keys unique -> all lanes converge)
  #pragma unroll
  for (int off = 16; off > 0; off >>= 1) {
    unsigned other[4];
    #pragma unroll
    for (int t = 0; t < 4; ++t)
      other[t] = (unsigned)__shfl_xor((int)best[t], off, 32);
    unsigned merged[4];
    int iq = 0, oq = 0;
    #pragma unroll
    for (int t = 0; t < 4; ++t) {
      bool takeA = (oq >= 4) | ((iq < 4) & (best[iq & 3] <= other[oq & 3]));
      merged[t] = takeA ? best[iq & 3] : other[oq & 3];
      iq += takeA ? 1 : 0;
      oq += takeA ? 0 : 1;
    }
    #pragma unroll
    for (int t = 0; t < 4; ++t) best[t] = merged[t];
  }

  float w[4];
  size_t jrow[4];
  #pragma unroll
  for (int t = 0; t < 4; ++t) {
    int j = (int)(best[t] & 2047u);
    jrow[t] = ((size_t)bb * NPTS + j) * DD;
    float d = sqrtf((float)(best[t] >> 11)) * (1.0f / 128.0f);
    w[t] = 0.5f - fminf(d, 0.5f);   // R - clip(dist,0,R)
  }

  const size_t rowA = (size_t)bb * NPTS + i;
  const float* fa  = feats_a + rowA * DD;
  const float* par = pa + rowA * DD;
  const float* q0 = pb + jrow[0];
  const float* q1 = pb + jrow[1];
  const float* q2 = pb + jrow[2];
  const float* q3 = pb + jrow[3];
  float* orow = out + rowA * (2 * DD);

  for (int d = lane; d < DD; d += 32) {
    orow[d] = fa[d];
    float av = par[d] + bias[d];
    float acc;
    acc  = w[0] * fmaxf(av + q0[d], 0.0f);
    acc += w[1] * fmaxf(av + q1[d], 0.0f);
    acc += w[2] * fmaxf(av + q2[d], 0.0f);
    acc += w[3] * fmaxf(av + q3[d], 0.0f);
    orow[DD + d] = acc;
  }
}

// ---------------------------------------------------------------------------
extern "C" void kernel_launch(void* const* d_in, const int* in_sizes, int n_in,
                              void* d_out, int out_size, void* d_ws, size_t ws_size,
                              hipStream_t stream) {
  const float* feats_a  = (const float*)d_in[0];
  const float* feats_b  = (const float*)d_in[1];
  const float* W        = (const float*)d_in[2];
  const float* bias     = (const float*)d_in[3];
  const int*   coords_a = (const int*)d_in[4];
  const int*   coords_b = (const int*)d_in[5];
  float* out = (float*)d_out;

  const size_t M = (size_t)NBAT * NPTS;          // 32768
  char* ws = (char*)d_ws;
  unsigned short* WbotT  = (unsigned short*)ws;                         // 512 KB
  unsigned short* WdiffT = (unsigned short*)(ws + (size_t)512 * 1024);  // 512 KB
  float* pa = (float*)(ws + (size_t)1024 * 1024);                      // 64 MB
  float* pb = (float*)(ws + (size_t)1024 * 1024 + M * DD * sizeof(float));

  prep_w<<<dim3((DD * DD) / 256), dim3(256), 0, stream>>>(W, WbotT, WdiffT);

  dim3 ggrid((unsigned)(M / 128), DD / 64);      // (256, 8)
  gemm_proj<<<ggrid, dim3(256), 0, stream>>>(feats_a, WbotT, pa);
  gemm_proj<<<ggrid, dim3(256), 0, stream>>>(feats_b, WdiffT, pb);

  fuse<<<dim3(NPTS / 8, NBAT), dim3(256), 0, stream>>>(
      feats_a, bias, coords_a, coords_b, pa, pb, out);
}
